// StyleLoss_37280316129817
// MI455X (gfx1250) — matrix operand order repaired and checked by reading, not compiled, not run on hardware
//
#include <hip/hip_runtime.h>

typedef __attribute__((ext_vector_type(16))) __bf16 v16bf;
typedef __attribute__((ext_vector_type(8)))  float  v8f;

#define WMMA_BF16(A, B, C) \
  __builtin_amdgcn_wmma_f32_16x16x32_bf16(false, (A), false, (B), (short)0, (C), false, false)

union Frag { v16bf v; uint4 u[2]; };

// ---------- helpers ----------
__device__ __forceinline__ unsigned f2key(float f) {
  unsigned u = __float_as_uint(f);
  return (u & 0x80000000u) ? ~u : (u | 0x80000000u);
}
__device__ __forceinline__ float key2f(unsigned k) {
  return (k & 0x80000000u) ? __uint_as_float(k & 0x7FFFFFFFu) : __uint_as_float(~k);
}
__device__ __forceinline__ unsigned short f2bf(float f) {
  unsigned u = __float_as_uint(f);
  u += 0x7FFFu + ((u >> 16) & 1u);           // round-to-nearest-even
  return (unsigned short)(u >> 16);
}
__device__ __forceinline__ unsigned pack2bf16(float a, float b) {
  return (unsigned)f2bf(a) | ((unsigned)f2bf(b) << 16);
}

// A fragment: 16x32 bf16, lane half h covers K = h*8..h*8+7 and 16+h*8..+7 (ISA 7.12.2)
__device__ __forceinline__ v16bf load_frag_a(const unsigned short* tile, int rowBase, int lane) {
  int h = lane >> 4, r = lane & 15;
  Frag f;
  const unsigned short* p = tile + (rowBase + r) * 32 + h * 8;
  f.u[0] = *(const uint4*)(p);
  f.u[1] = *(const uint4*)(p + 16);
  return f.v;
}
// B fragment: 32x16 bf16, lane half h covers K = h*8..h*8+7 and 16+h*8..+7 (ISA 7.12.2)
__device__ __forceinline__ v16bf load_frag_b(const unsigned short* tile, int rowBase, int lane) {
  int h = lane >> 4, r = lane & 15;
  Frag f;
  const unsigned short* p = tile + (rowBase + r) * 32 + h * 8;
  f.u[0] = *(const uint4*)(p);
  f.u[1] = *(const uint4*)(p + 16);
  return f.v;
}

// ---------- init ----------
__global__ __launch_bounds__(256) void init_kernel(
    float* out, unsigned* cm32, float* cs32, unsigned* rm32,
    unsigned* cm42, float* cs42, unsigned* rm42) {
  int i = blockIdx.x * 256 + threadIdx.x;
  if (i < 2) out[i] = 0.0f;
  if (i < 4 * 4096) { cm32[i] = 0xFFFFFFFFu; cs32[i] = 0.0f; rm32[i] = 0u; }
  if (i < 4 * 1024) { cm42[i] = 0xFFFFFFFFu; cs42[i] = 0.0f; rm42[i] = 0u; }
}

// ---------- fused Gram + L1 ----------
// G = F F^T / (N*C); accumulate scale*sum|Gx-Gy| into out_style (scale folds mean).
// grid: (C/64 j, C/64 i, B); block 256 = 8 waves -> 4(i)x2(j), wave tile 16x32.
__global__ __launch_bounds__(256) void gram_l1_kernel(
    const float* __restrict__ X, const float* __restrict__ Y,
    float* out_style, int C, int N, float scale) {
  __shared__ __align__(16) unsigned short S[4 * 2048];  // Xi,Xj,Yi,Yj tiles 64x32 bf16
  int b = blockIdx.z;
  int i0 = blockIdx.y * 64, j0 = blockIdx.x * 64;
  const float* Xb = X + (size_t)b * C * N;
  const float* Yb = Y + (size_t)b * C * N;
  int t = threadIdx.x, lane = t & 31, wave = t >> 5;
  int wi = wave & 3, wj = wave >> 2;

  v8f z8 = {0.f, 0.f, 0.f, 0.f, 0.f, 0.f, 0.f, 0.f};
  v8f gx[2] = {z8, z8}, gy[2] = {z8, z8};

  for (int k0 = 0; k0 < N; k0 += 32) {
    #pragma unroll
    for (int i = 0; i < 8; ++i) {              // 2048 float4-chunks across 256 threads
      int cid = t + i * 256;
      int tile = cid >> 9;
      int rem = cid & 511;
      int row = rem >> 3;
      int co = (rem & 7) * 4;
      const float* src = (tile & 2) ? Yb : Xb;
      int grow = (tile & 1) ? (j0 + row) : (i0 + row);
      float4 v = *(const float4*)(src + (size_t)grow * N + k0 + co);
      uint2 pv; pv.x = pack2bf16(v.x, v.y); pv.y = pack2bf16(v.z, v.w);
      *(uint2*)(&S[tile * 2048 + row * 32 + co]) = pv;
    }
    __syncthreads();
    v16bf ax  = load_frag_a(S + 0 * 2048, wi * 16, lane);
    v16bf ay  = load_frag_a(S + 2 * 2048, wi * 16, lane);
    v16bf bx0 = load_frag_b(S + 1 * 2048, wj * 32,      lane);
    v16bf bx1 = load_frag_b(S + 1 * 2048, wj * 32 + 16, lane);
    v16bf by0 = load_frag_b(S + 3 * 2048, wj * 32,      lane);
    v16bf by1 = load_frag_b(S + 3 * 2048, wj * 32 + 16, lane);
    gx[0] = WMMA_BF16(ax, bx0, gx[0]);
    gx[1] = WMMA_BF16(ax, bx1, gx[1]);
    gy[0] = WMMA_BF16(ay, by0, gy[0]);
    gy[1] = WMMA_BF16(ay, by1, gy[1]);
    __syncthreads();
  }

  float s = 0.0f;
  #pragma unroll
  for (int tj = 0; tj < 2; ++tj)
    #pragma unroll
    for (int c = 0; c < 8; ++c)
      s += fabsf(gx[tj][c] - gy[tj][c]);
  s *= scale;
  #pragma unroll
  for (int m = 1; m < 32; m <<= 1) s += __shfl_xor(s, m, 32);
  if (lane == 0) atomicAdd(out_style, s);
}

// ---------- MRF normalize: write Tn^T, Gn^T as [b][p][c] bf16 ----------
__global__ __launch_bounds__(256) void mrf_normalize_kernel(
    const float* __restrict__ gen, const float* __restrict__ tar,
    unsigned short* TnT, unsigned short* GnT, int C, int P) {
  int b = blockIdx.y;
  int q = blockIdx.x * 256 + threadIdx.x;
  const float* g  = gen + (size_t)b * C * P + q;
  const float* ta = tar + (size_t)b * C * P + q;
  float mean = 0.0f;
  for (int c = 0; c < C; ++c) mean += ta[(size_t)c * P];
  mean /= (float)C;
  float sg = 0.0f, st = 0.0f;
  for (int c = 0; c < C; ++c) {
    float gv = g[(size_t)c * P] - mean;
    float tv = ta[(size_t)c * P] - mean;
    sg += gv * gv; st += tv * tv;
  }
  float ig = rsqrtf(sg), it = rsqrtf(st);
  unsigned short* to = TnT + ((size_t)b * P + q) * C;
  unsigned short* go = GnT + ((size_t)b * P + q) * C;
  for (int c = 0; c < C; ++c) {
    to[c] = f2bf((ta[(size_t)c * P] - mean) * it);
    go[c] = f2bf((g[(size_t)c * P] - mean) * ig);
  }
}

// ---------- MRF GEMM passes: 0=col-min(cdist), 1=col-sum(e), 2=row-max(cs) ----------
// cos[p,q] = sum_c TnT[p,c]*GnT[q,c]; block tile 128(p)x64(q); waves 4(p)x2(q), 32x32 each.
template <int PASS>
__global__ __launch_bounds__(256) void mrf_pass_kernel(
    const unsigned short* __restrict__ TnT, const unsigned short* __restrict__ GnT,
    unsigned* colmin, float* colsum, unsigned* rowmax, int C, int P) {
  __shared__ __align__(16) unsigned short As[128 * 32];
  __shared__ __align__(16) unsigned short Bs[64 * 32];
  __shared__ float red[256];
  int b = blockIdx.z;
  int p0 = blockIdx.y * 128;
  int q0 = blockIdx.x * 64;
  int t = threadIdx.x, lane = t & 31, wave = t >> 5;
  int wi = wave & 3, wj = wave >> 2;
  const unsigned short* Abase = TnT + ((size_t)b * P + p0) * C;
  const unsigned short* Bbase = GnT + ((size_t)b * P + q0) * C;

  v8f z8 = {0.f, 0.f, 0.f, 0.f, 0.f, 0.f, 0.f, 0.f};
  v8f acc[2][2] = {{z8, z8}, {z8, z8}};

  for (int k0 = 0; k0 < C; k0 += 32) {
    #pragma unroll
    for (int i = 0; i < 2; ++i) {             // A: 128 rows x 4 chunks
      int chunk = t + i * 256;
      int row = chunk >> 2;
      int co = (chunk & 3) * 8;
      *(uint4*)(&As[row * 32 + co]) = *(const uint4*)(Abase + (size_t)row * C + k0 + co);
    }
    {                                          // B: 64 rows x 4 chunks
      int row = t >> 2;
      int co = (t & 3) * 8;
      *(uint4*)(&Bs[row * 32 + co]) = *(const uint4*)(Bbase + (size_t)row * C + k0 + co);
    }
    __syncthreads();
    v16bf a0 = load_frag_a(As, wi * 32,      lane);
    v16bf a1 = load_frag_a(As, wi * 32 + 16, lane);
    v16bf b0 = load_frag_b(Bs, wj * 32,      lane);
    v16bf b1 = load_frag_b(Bs, wj * 32 + 16, lane);
    acc[0][0] = WMMA_BF16(a0, b0, acc[0][0]);
    acc[0][1] = WMMA_BF16(a0, b1, acc[0][1]);
    acc[1][0] = WMMA_BF16(a1, b0, acc[1][0]);
    acc[1][1] = WMMA_BF16(a1, b1, acc[1][1]);
    __syncthreads();
  }

  if constexpr (PASS == 0) {
    #pragma unroll
    for (int tj = 0; tj < 2; ++tj) {
      float m = 1e30f;
      #pragma unroll
      for (int ti = 0; ti < 2; ++ti)
        #pragma unroll
        for (int c = 0; c < 8; ++c)
          m = fminf(m, 0.5f * (1.0f - acc[ti][tj][c]));
      m = fminf(m, __shfl_xor(m, 16, 32));
      if (lane < 16) red[wi * 64 + wj * 32 + tj * 16 + lane] = m;
    }
    __syncthreads();
    if (t < 64) {
      float m = fminf(fminf(red[t], red[64 + t]), fminf(red[128 + t], red[192 + t]));
      atomicMin(&colmin[(size_t)b * P + q0 + t], f2key(m));
    }
  } else {
    float mq[2];
    #pragma unroll
    for (int tj = 0; tj < 2; ++tj)
      mq[tj] = key2f(colmin[(size_t)b * P + q0 + wj * 32 + tj * 16 + (lane & 15)]) + 1e-5f;

    if constexpr (PASS == 1) {
      #pragma unroll
      for (int tj = 0; tj < 2; ++tj) {
        float s = 0.0f;
        #pragma unroll
        for (int ti = 0; ti < 2; ++ti)
          #pragma unroll
          for (int c = 0; c < 8; ++c) {
            float cd = 0.5f * (1.0f - acc[ti][tj][c]);
            s += __expf((1.0f - cd / mq[tj]) * 2.0f);   // (BIAS-rel)/SIGMA
          }
        s += __shfl_xor(s, 16, 32);
        if (lane < 16) red[wi * 64 + wj * 32 + tj * 16 + lane] = s;
      }
      __syncthreads();
      if (t < 64) {
        float s = red[t] + red[64 + t] + red[128 + t] + red[192 + t];
        atomicAdd(&colsum[(size_t)b * P + q0 + t], s);
      }
    } else {  // PASS 2: row max of cs = e / colsum
      float sq[2];
      #pragma unroll
      for (int tj = 0; tj < 2; ++tj)
        sq[tj] = colsum[(size_t)b * P + q0 + wj * 32 + tj * 16 + (lane & 15)];
      int h = lane >> 4;
      #pragma unroll
      for (int ti = 0; ti < 2; ++ti)
        #pragma unroll
        for (int c = 0; c < 8; ++c) {
          float cd0 = 0.5f * (1.0f - acc[ti][0][c]);
          float cd1 = 0.5f * (1.0f - acc[ti][1][c]);
          float v0 = __expf((1.0f - cd0 / mq[0]) * 2.0f) / sq[0];
          float v1 = __expf((1.0f - cd1 / mq[1]) * 2.0f) / sq[1];
          float v = fmaxf(v0, v1);
          v = fmaxf(v, __shfl_xor(v, 1, 32));
          v = fmaxf(v, __shfl_xor(v, 2, 32));
          v = fmaxf(v, __shfl_xor(v, 4, 32));
          v = fmaxf(v, __shfl_xor(v, 8, 32));
          if ((lane & 15) == 0) red[wj * 128 + wi * 32 + ti * 16 + h * 8 + c] = v;
        }
      __syncthreads();
      if (t < 128) {
        float v = fmaxf(red[t], red[128 + t]);
        atomicMax(&rowmax[(size_t)b * P + p0 + t], f2key(v));
      }
    }
  }
}

// ---------- final: mrf = sum_b -log(mean_p rowmax) per layer (x42 weight 2) ----------
__global__ __launch_bounds__(256) void mrf_final_kernel(
    const unsigned* __restrict__ rm32, const unsigned* __restrict__ rm42, float* out) {
  __shared__ float red[256];
  int t = threadIdx.x;
  float total = 0.0f;
  for (int layer = 0; layer < 2; ++layer) {
    const unsigned* rm = layer ? rm42 : rm32;
    int P = layer ? 1024 : 4096;
    float w = layer ? 2.0f : 1.0f;
    for (int b = 0; b < 4; ++b) {
      float s = 0.0f;
      for (int i = t; i < P; i += 256) s += key2f(rm[(size_t)b * P + i]);
      red[t] = s;
      __syncthreads();
      for (int m = 128; m > 0; m >>= 1) {
        if (t < m) red[t] += red[t + m];
        __syncthreads();
      }
      if (t == 0) total += -logf(red[0] / (float)P) * w;
      __syncthreads();
    }
  }
  if (t == 0) out[1] = total;
}

static inline float gram_scale(int C, int N) {
  return (float)(1.0 / ((double)N * (double)C * 4.0 * (double)C * (double)C));
}

extern "C" void kernel_launch(void* const* d_in, const int* in_sizes, int n_in,
                              void* d_out, int out_size, void* d_ws, size_t ws_size,
                              hipStream_t stream) {
  (void)in_sizes; (void)n_in; (void)out_size; (void)ws_size;
  const float* x22 = (const float*)d_in[0];
  const float* y22 = (const float*)d_in[1];
  const float* x34 = (const float*)d_in[2];
  const float* y34 = (const float*)d_in[3];
  const float* x44 = (const float*)d_in[4];
  const float* y44 = (const float*)d_in[5];
  const float* x52 = (const float*)d_in[6];
  const float* y52 = (const float*)d_in[7];
  const float* x32 = (const float*)d_in[8];
  const float* y32 = (const float*)d_in[9];
  const float* x42 = (const float*)d_in[10];
  const float* y42 = (const float*)d_in[11];
  float* out = (float*)d_out;

  char* ws = (char*)d_ws;
  size_t off = 0;
  auto carve = [&](size_t bytes) -> void* {
    void* p = ws + off;
    off += (bytes + 255) & ~(size_t)255;
    return p;
  };
  unsigned short* TnT32 = (unsigned short*)carve((size_t)4 * 4096 * 256 * 2);
  unsigned short* GnT32 = (unsigned short*)carve((size_t)4 * 4096 * 256 * 2);
  unsigned short* TnT42 = (unsigned short*)carve((size_t)4 * 1024 * 512 * 2);
  unsigned short* GnT42 = (unsigned short*)carve((size_t)4 * 1024 * 512 * 2);
  unsigned* cm32 = (unsigned*)carve((size_t)4 * 4096 * 4);
  float*    cs32 = (float*)carve((size_t)4 * 4096 * 4);
  unsigned* rm32 = (unsigned*)carve((size_t)4 * 4096 * 4);
  unsigned* cm42 = (unsigned*)carve((size_t)4 * 1024 * 4);
  float*    cs42 = (float*)carve((size_t)4 * 1024 * 4);
  unsigned* rm42 = (unsigned*)carve((size_t)4 * 1024 * 4);

  init_kernel<<<64, 256, 0, stream>>>(out, cm32, cs32, rm32, cm42, cs42, rm42);

  // Style loss (atomicAdd into out[0])
  gram_l1_kernel<<<dim3(2, 2, 4), 256, 0, stream>>>(x22, y22, out, 128, 16384, gram_scale(128, 16384));
  gram_l1_kernel<<<dim3(4, 4, 4), 256, 0, stream>>>(x34, y34, out, 256, 4096,  gram_scale(256, 4096));
  gram_l1_kernel<<<dim3(8, 8, 4), 256, 0, stream>>>(x44, y44, out, 512, 1024,  gram_scale(512, 1024));
  gram_l1_kernel<<<dim3(8, 8, 4), 256, 0, stream>>>(x52, y52, out, 512, 256,   gram_scale(512, 256));

  // MRF layer 32: gen=x32, tar=y32, C=256, P=4096
  mrf_normalize_kernel<<<dim3(16, 4), 256, 0, stream>>>(x32, y32, TnT32, GnT32, 256, 4096);
  // MRF layer 42: gen=x42, tar=y42, C=512, P=1024
  mrf_normalize_kernel<<<dim3(4, 4), 256, 0, stream>>>(x42, y42, TnT42, GnT42, 512, 1024);

  dim3 g32(4096 / 64, 4096 / 128, 4);
  dim3 g42(1024 / 64, 1024 / 128, 4);
  mrf_pass_kernel<0><<<g32, 256, 0, stream>>>(TnT32, GnT32, cm32, cs32, rm32, 256, 4096);
  mrf_pass_kernel<0><<<g42, 256, 0, stream>>>(TnT42, GnT42, cm42, cs42, rm42, 512, 1024);
  mrf_pass_kernel<1><<<g32, 256, 0, stream>>>(TnT32, GnT32, cm32, cs32, rm32, 256, 4096);
  mrf_pass_kernel<1><<<g42, 256, 0, stream>>>(TnT42, GnT42, cm42, cs42, rm42, 512, 1024);
  mrf_pass_kernel<2><<<g32, 256, 0, stream>>>(TnT32, GnT32, cm32, cs32, rm32, 256, 4096);
  mrf_pass_kernel<2><<<g42, 256, 0, stream>>>(TnT42, GnT42, cm42, cs42, rm42, 512, 1024);

  mrf_final_kernel<<<1, 256, 0, stream>>>(rm32, rm42, out);
}
